// DisentangledSpatialSA_11622181503259
// MI455X (gfx1250) — matrix operand order repaired and checked by reading, not compiled, not run on hardware
//
#include <hip/hip_runtime.h>
#include <hip/hip_bf16.h>

// ---------------------------------------------------------------------------
// DisentangledSpatialSA for MI455X (gfx1250, wave32, WMMA 16x16x32 bf16)
// Flash-attention formulation: never materialize the 4096x4096 attention
// matrix in HBM (saves ~1.6 GB of traffic @ 23.3 TB/s).
// Round 3: GLOBAL_LOAD_ASYNC_TO_LDS_B128 (ASYNCcnt DMA) + double-buffered
// LDS tiles so the next j-tile streams in while WMMAs consume the current.
// ---------------------------------------------------------------------------

#define B_SZ   8
#define CH_SZ  256
#define IC_SZ  128
#define N_SP   4096          // 64*64
#define O_SZ   384           // 3*IC
#define ALPHA  1.76776695f   // 1/(sqrt(128)*0.05)

typedef __bf16 bf16_t;
typedef bf16_t v16bf __attribute__((ext_vector_type(16)));
typedef float  v8f   __attribute__((ext_vector_type(8)));

union BF16x16 { v16bf v; bf16_t h[16]; uint4 u4[2]; };

__device__ __forceinline__ void split_bf16(float f, bf16_t& hi, bf16_t& lo) {
    hi = (bf16_t)f;
    lo = (bf16_t)(f - (float)hi);
}
__device__ __forceinline__ v8f wmma_bf16(const BF16x16& a, const BF16x16& b, v8f c) {
    return __builtin_amdgcn_wmma_f32_16x16x32_bf16(false, a.v, false, b.v,
                                                   (short)0, c, false, false);
}

// Async DMA: 16 bytes global -> LDS per lane, tracked by ASYNCcnt.
// VDST carries the LDS byte address (low 32 bits of the generic pointer),
// VADDR the 64-bit global address (GV mode, saddr = off).
__device__ __forceinline__ void async_copy16(void* lds_dst, const void* gsrc) {
    unsigned lds = (unsigned)(size_t)lds_dst;
    asm volatile("global_load_async_to_lds_b128 %0, %1, off"
                 :: "v"(lds), "v"(gsrc) : "memory");
}
__device__ __forceinline__ void wait_async0() {
    asm volatile("s_wait_asynccnt 0x0" ::: "memory");
}

// ---------------------------------------------------------------------------
// Kernel 1: xmean[b][c] = mean over 4096 spatial positions (exploits
// linearity: mean(q) = w_q @ mean(x) + b, so q/k centering folds into proj)
// ---------------------------------------------------------------------------
__global__ __launch_bounds__(256) void reduce_xmean(const float* __restrict__ x,
                                                    float* __restrict__ xmean) {
    const int bc = blockIdx.x;                 // 0 .. B*CH-1
    const float* row = x + (size_t)bc * N_SP;
    float s = 0.f;
    for (int i = threadIdx.x; i < N_SP; i += 256) s += row[i];
    __shared__ float red[256];
    red[threadIdx.x] = s;
    __syncthreads();
    for (int st = 128; st > 0; st >>= 1) {
        if (threadIdx.x < st) red[threadIdx.x] += red[threadIdx.x + st];
        __syncthreads();
    }
    if (threadIdx.x == 0) xmean[bc] = red[0] * (1.f / N_SP);
}

// ---------------------------------------------------------------------------
// Kernel 2: mdot[b][o] = w_qkv[o,:] @ xmean[b,:]   (tiny, 3072 dots of 256)
// ---------------------------------------------------------------------------
__global__ __launch_bounds__(256) void compute_mdot(const float* __restrict__ w_qkv,
                                                    const float* __restrict__ xmean,
                                                    float* __restrict__ mdot) {
    int gid = blockIdx.x * 256 + threadIdx.x;
    if (gid >= B_SZ * O_SZ) return;
    int b = gid / O_SZ, o = gid % O_SZ;
    const float* wr = w_qkv + (size_t)o * CH_SZ;
    const float* xm = xmean + b * CH_SZ;
    float s = 0.f;
    for (int k = 0; k < CH_SZ; ++k) s += wr[k] * xm[k];
    mdot[gid] = s;
}

// ---------------------------------------------------------------------------
// Kernel 3: QKV projection, split-bf16 WMMA (hi*hi + hi*lo + lo*hi ~ fp32).
// Writes centered q,k transposed as qT/kT [b][n][128] bf16 (this is exactly
// the contiguous store pattern of the f32 C-fragment layout), and v as
// [b][128][n] bf16.
// grid = (32 n-tiles, 8 batches), 256 threads = 8 waves.
// ---------------------------------------------------------------------------
__global__ __launch_bounds__(256) void qkv_proj(const float* __restrict__ x,
                                                const float* __restrict__ w_qkv,
                                                const float* __restrict__ b_qkv,
                                                const float* __restrict__ mdot,
                                                bf16_t* __restrict__ qT,
                                                bf16_t* __restrict__ kT,
                                                bf16_t* __restrict__ vM) {
    const int b   = blockIdx.y;
    const int n0  = blockIdx.x * 128;
    const int tid = threadIdx.x;
    const int w   = tid >> 5;
    const int l   = tid & 31;
    const int l15 = l & 15;
    const int lh  = l >> 4;            // half-wave id

    // LDS: x chunk transposed to [n_local][k_local], bf16 hi/lo, stride 40
    __shared__ __align__(16) bf16_t xhi[128 * 40];
    __shared__ __align__(16) bf16_t xlo[128 * 40];

    const float* xb = x + (size_t)b * CH_SZ * N_SP;

    for (int t = 0; t < 3; ++t) {                   // t=0:q  t=1:k  t=2:v
        const int mb = w + 8 * t;                   // m-block (16 output rows)
        v8f acc[8];
        for (int nb = 0; nb < 8; ++nb)
            for (int r = 0; r < 8; ++r) acc[nb][r] = 0.f;

        for (int kc = 0; kc < 8; ++kc) {            // K chunks of 32
            __syncthreads();
            {   // cooperative stage: x[kc*32 .. +31][n0 .. n0+127] -> LDS^T
                int r  = tid >> 3;                  // 0..31 (k local)
                int c0 = (tid & 7) * 16;            // 0..112 (n local)
                const float* src = xb + (size_t)(kc * 32 + r) * N_SP + n0 + c0;
                for (int i = 0; i < 16; ++i) {
                    bf16_t h, lo;
                    split_bf16(src[i], h, lo);
                    xhi[(c0 + i) * 40 + r] = h;
                    xlo[(c0 + i) * 40 + r] = lo;
                }
            }
            __syncthreads();

            // A fragment from w_qkv (fp32 global, on-the-fly hi/lo split)
            BF16x16 ahi, alo;
            {
                int o = mb * 16 + l15;
                const float* wr = w_qkv + (size_t)o * CH_SZ + kc * 32 + lh * 16;
                for (int i = 0; i < 8; ++i) {
                    split_bf16(wr[i],      ahi.h[i],     alo.h[i]);
                    split_bf16(wr[i + 8],  ahi.h[i + 8], alo.h[i + 8]);
                }
            }

            const uint4* xh4 = (const uint4*)xhi;
            const uint4* xl4 = (const uint4*)xlo;
            for (int nb = 0; nb < 8; ++nb) {
                BF16x16 bh, bl;
                int n   = nb * 16 + l15;
                int idx = n * 5 + lh * 2;           // 40 halves = 5 uint4/row
                bh.u4[0] = xh4[idx]; bh.u4[1] = xh4[idx + 1];
                bl.u4[0] = xl4[idx]; bl.u4[1] = xl4[idx + 1];
                acc[nb] = wmma_bf16(ahi, bh, acc[nb]);
                acc[nb] = wmma_bf16(ahi, bl, acc[nb]);
                acc[nb] = wmma_bf16(alo, bh, acc[nb]);
            }
        }

        // Epilogue: center q/k (subtract w@xmean), add bias for v, store bf16
        if (t < 2) {
            bf16_t* dstT = (t == 0) ? qT : kT;
            uint4* dT = (uint4*)(dstT + (size_t)b * N_SP * IC_SZ);
            int mloc = (t == 0) ? mb : (mb - 8);
            for (int nb = 0; nb < 8; ++nb) {
                int col = n0 + nb * 16 + l15;
                BF16x16 pk;
                for (int r = 0; r < 8; ++r) {
                    int o = mb * 16 + lh * 8 + r;   // global qkv row
                    pk.h[r] = (bf16_t)(acc[nb][r] - mdot[b * O_SZ + o]);
                }
                dT[col * 16 + mloc * 2 + lh] = pk.u4[0];  // 8 contiguous ch
            }
        } else {
            bf16_t* vb = vM + (size_t)b * IC_SZ * N_SP;
            for (int nb = 0; nb < 8; ++nb) {
                int col = n0 + nb * 16 + l15;
                for (int r = 0; r < 8; ++r) {
                    int o = mb * 16 + lh * 8 + r;   // 256..383
                    vb[(size_t)(o - 256) * N_SP + col] =
                        (bf16_t)(acc[nb][r] + b_qkv[o]);
                }
            }
        }
    }
}

// ---------------------------------------------------------------------------
// Kernel 4: flash attention. S^T[j,i] = sum_c q[c,j]k[c,i] so softmax axis (j)
// is the WMMA M axis: per-column reduce = 8 in-lane values + 1 shfl_xor(16).
// Each wave owns 16 i-columns; workgroup owns 128 i; loop over j in tiles of
// 128. Tiles stream into double-buffered LDS via GLOBAL_LOAD_ASYNC_TO_LDS
// (ASYNCcnt DMA) overlapping the 64 WMMAs of the current tile.
// grid = (32 i-tiles, 8 batches), 256 threads = 8 waves.
// ---------------------------------------------------------------------------
__global__ __launch_bounds__(256) void flash_attn(const bf16_t* __restrict__ qT,
                                                  const bf16_t* __restrict__ kT,
                                                  const bf16_t* __restrict__ vM,
                                                  bf16_t* __restrict__ yT) {
    const int b   = blockIdx.y;
    const int i0  = blockIdx.x * 128;
    const int tid = threadIdx.x;
    const int w   = tid >> 5;
    const int l   = tid & 31;
    const int l15 = l & 15;
    const int lh  = l >> 4;

    // Double-buffered LDS tiles, stride 136 halves = 17 uint4 (bank padding)
    __shared__ uint4 qs[2][128 * 17];  // Q tile  [j_local][c]
    __shared__ uint4 vs[2][128 * 17];  // V tile  [c][j_local]

    const uint4* qTg = (const uint4*)(qT + (size_t)b * N_SP * IC_SZ);
    const uint4* kTg = (const uint4*)(kT + (size_t)b * N_SP * IC_SZ);
    const uint4* vg4 = (const uint4*)(vM + (size_t)b * IC_SZ * N_SP);

    // issue async DMA of tile `jt` into buffer `buf` (8 x 2 per thread)
    auto stage_async = [&](int jt, int buf) {
        for (int f = tid; f < 2048; f += 256) {
            int row = f >> 4, c16 = f & 15;
            async_copy16(&qs[buf][row * 17 + c16],
                         &qTg[(jt * 128 + row) * 16 + c16]);
            async_copy16(&vs[buf][row * 17 + c16],
                         &vg4[row * (N_SP / 8) + jt * 16 + c16]);
        }
    };

    // K B-fragments for this wave's 16 i-columns (column-contiguous from kT)
    BF16x16 bk[4];
    {
        int i = i0 + w * 16 + l15;
        for (int cc = 0; cc < 4; ++cc) {
            int base = i * 16 + cc * 4 + lh * 2;
            bk[cc].u4[0] = kTg[base];
            bk[cc].u4[1] = kTg[base + 1];
        }
    }

    float m_run = -1e30f, l_run = 0.f;
    v8f yacc[8];
    for (int cb = 0; cb < 8; ++cb)
        for (int r = 0; r < 8; ++r) yacc[cb][r] = 0.f;

    // prologue: fill buffer 0
    stage_async(0, 0);
    wait_async0();
    __syncthreads();

    for (int jt = 0; jt < 32; ++jt) {
        const int cur = jt & 1;
        // kick off DMA of the next tile into the other buffer
        if (jt + 1 < 32) stage_async(jt + 1, cur ^ 1);

        const uint4* qcur = qs[cur];
        const uint4* vcur = vs[cur];

        // S^T blocks: M=j (8 blocks), N = wave's 16 i-columns, K=c (4 chunks)
        v8f s[8];
        for (int jb = 0; jb < 8; ++jb) {
            v8f a;
            for (int r = 0; r < 8; ++r) a[r] = 0.f;
            int row = jb * 16 + l15;
            for (int cc = 0; cc < 4; ++cc) {
                BF16x16 aq;
                int idx = row * 17 + cc * 4 + lh * 2;
                aq.u4[0] = qcur[idx];
                aq.u4[1] = qcur[idx + 1];
                a = wmma_bf16(aq, bk[cc], a);
            }
            s[jb] = a;
        }

        // online softmax over j (per i-column = per lane, + half-wave pair)
        float tmax = -1e30f;
        for (int jb = 0; jb < 8; ++jb)
            for (int r = 0; r < 8; ++r) {
                float v = s[jb][r] * ALPHA;
                s[jb][r] = v;
                tmax = fmaxf(tmax, v);
            }
        tmax = fmaxf(tmax, __shfl_xor(tmax, 16, 32));
        float m_new = fmaxf(m_run, tmax);
        float corr  = __expf(m_run - m_new);
        float tsum = 0.f;
        for (int jb = 0; jb < 8; ++jb)
            for (int r = 0; r < 8; ++r) {
                float p = __expf(s[jb][r] - m_new);
                s[jb][r] = p;
                tsum += p;
            }
        tsum += __shfl_xor(tsum, 16, 32);
        l_run = l_run * corr + tsum;
        m_run = m_new;
        for (int cb = 0; cb < 8; ++cb)
            for (int r = 0; r < 8; ++r) yacc[cb][r] *= corr;

        // P^T : C-layout -> B-layout bf16 (half-wave exchange)
        BF16x16 bp[4];
        for (int jc = 0; jc < 4; ++jc) {
            for (int r = 0; r < 8; ++r) {
                float o0 = __shfl_xor(s[2 * jc][r], 16, 32);
                float o1 = __shfl_xor(s[2 * jc + 1][r], 16, 32);
                if (lh == 0) {
                    bp[jc].h[r]     = (bf16_t)s[2 * jc][r];
                    bp[jc].h[r + 8] = (bf16_t)o0;
                } else {
                    bp[jc].h[r]     = (bf16_t)o1;
                    bp[jc].h[r + 8] = (bf16_t)s[2 * jc + 1][r];
                }
            }
        }

        // y[c, i] += V_A(c x j) @ P^T_B(j x i)
        for (int cb = 0; cb < 8; ++cb) {
            int row = cb * 16 + l15;
            for (int jc = 0; jc < 4; ++jc) {
                BF16x16 av;
                int idx = row * 17 + jc * 4 + lh * 2;
                av.u4[0] = vcur[idx];
                av.u4[1] = vcur[idx + 1];
                yacc[cb] = wmma_bf16(av, bp[jc], yacc[cb]);
            }
        }

        // next tile's DMA must have landed; all waves done with buffers
        wait_async0();
        __syncthreads();
    }

    // finalize: y /= l, store transposed yT[b][i][c] (contiguous C-frag store)
    float inv = 1.f / l_run;
    int i = i0 + w * 16 + l15;
    uint4* yTg = (uint4*)(yT + (size_t)b * N_SP * IC_SZ);
    for (int cb = 0; cb < 8; ++cb) {
        BF16x16 pk;
        for (int r = 0; r < 8; ++r) pk.h[r] = (bf16_t)(yacc[cb][r] * inv);
        yTg[i * 16 + cb * 2 + lh] = pk.u4[0];
    }
}

// ---------------------------------------------------------------------------
// Kernel 5: out = x + w_out @ y + b_out, split-bf16 WMMA for w_out.
// grid = (32 n-tiles, 8 batches), 256 threads = 8 waves.
// ---------------------------------------------------------------------------
__global__ __launch_bounds__(256) void out_proj(const float* __restrict__ x,
                                                const float* __restrict__ w_out,
                                                const float* __restrict__ b_out,
                                                const bf16_t* __restrict__ yT,
                                                float* __restrict__ out) {
    const int b   = blockIdx.y;
    const int n0  = blockIdx.x * 128;
    const int tid = threadIdx.x;
    const int w   = tid >> 5;
    const int l   = tid & 31;
    const int l15 = l & 15;
    const int lh  = l >> 4;

    const uint4* yg = (const uint4*)(yT + (size_t)b * N_SP * IC_SZ);
    const float* xb = x + (size_t)b * CH_SZ * N_SP;
    float* ob = out + (size_t)b * CH_SZ * N_SP;

    for (int t = 0; t < 2; ++t) {
        const int mb = w + 8 * t;                   // 16 m-blocks total
        BF16x16 ahi[4], alo[4];
        {
            int o = mb * 16 + l15;
            for (int cc = 0; cc < 4; ++cc) {
                const float* wr = w_out + (size_t)o * IC_SZ + cc * 32 + lh * 16;
                for (int i = 0; i < 8; ++i) {
                    split_bf16(wr[i],      ahi[cc].h[i],     alo[cc].h[i]);
                    split_bf16(wr[i + 8],  ahi[cc].h[i + 8], alo[cc].h[i + 8]);
                }
            }
        }
        for (int nb = 0; nb < 8; ++nb) {
            v8f acc;
            for (int r = 0; r < 8; ++r) acc[r] = 0.f;
            int n = n0 + nb * 16 + l15;
            for (int cc = 0; cc < 4; ++cc) {
                BF16x16 bv;
                int idx = n * 16 + cc * 4 + lh * 2;
                bv.u4[0] = yg[idx];
                bv.u4[1] = yg[idx + 1];
                acc = wmma_bf16(ahi[cc], bv, acc);
                acc = wmma_bf16(alo[cc], bv, acc);
            }
            for (int r = 0; r < 8; ++r) {
                int o = mb * 16 + lh * 8 + r;
                size_t off = (size_t)o * N_SP + n;
                ob[off] = xb[off] + acc[r] + b_out[o];
            }
        }
    }
}

// ---------------------------------------------------------------------------
extern "C" void kernel_launch(void* const* d_in, const int* in_sizes, int n_in,
                              void* d_out, int out_size, void* d_ws, size_t ws_size,
                              hipStream_t stream) {
    const float* x      = (const float*)d_in[0];
    const float* w_qkv  = (const float*)d_in[1];
    const float* b_qkv  = (const float*)d_in[2];
    const float* w_out  = (const float*)d_in[3];
    const float* b_out  = (const float*)d_in[4];
    float* out = (float*)d_out;

    char* ws = (char*)d_ws;
    const size_t HALF_CNT = (size_t)B_SZ * N_SP * IC_SZ;    // 4.19M halves
    float*  xmean = (float*)ws;                              // 8 KB
    float*  mdot  = (float*)(ws + 16384);                    // 12 KB
    bf16_t* qT    = (bf16_t*)(ws + 32768);                   // 8 MB each
    bf16_t* kT    = qT + HALF_CNT;
    bf16_t* vM    = kT + HALF_CNT;
    bf16_t* yT    = vM + HALF_CNT;

    reduce_xmean<<<B_SZ * CH_SZ, 256, 0, stream>>>(x, xmean);
    compute_mdot<<<(B_SZ * O_SZ + 255) / 256, 256, 0, stream>>>(w_qkv, xmean, mdot);
    qkv_proj<<<dim3(32, B_SZ), 256, 0, stream>>>(x, w_qkv, b_qkv, mdot, qT, kT, vM);
    flash_attn<<<dim3(32, B_SZ), 256, 0, stream>>>(qT, kT, vM, yT);
    out_proj<<<dim3(32, B_SZ), 256, 0, stream>>>(x, w_out, b_out, yT, out);
}
